// TreeAggregatorCell_80556406604249
// MI455X (gfx1250) — compile-verified, matrix operand order checked by reading, not run on hardware
//
#include <hip/hip_runtime.h>
#include <cstdint>
#include <cstddef>

// ---------------------------------------------------------------------------
// TreeAggregatorCell fused kernel for gfx1250 (MI455X), wave32 + WMMA bf16.
// N=20000 nodes, K=16 mail slots, H=X=256.
// Prep kernel pre-swizzles all weights into WMMA B-fragment bf16 layout in ws.
// Main kernel: one workgroup (8 waves) per 16-node tile; wave owns 32 columns.
// Double-buffered h_m staging -> one barrier per mail slot.
// ---------------------------------------------------------------------------

#define NNODES 20000
#define KMAIL  16
#define HDIM   256
#define XDIM   256
#define TILE_M 16
#define NTHREADS 256

// ws layout (dwords): per 16-col tile: 8 ksteps * 32 lanes * 8 dwords = 2048
#define TILE_DW   2048
#define WS_UF_DW  0                 // U_f   : 16 tiles
#define WS_WF_DW  (16 * TILE_DW)    // W_f   : 16 tiles
#define WS_UIOU_DW (32 * TILE_DW)   // U_iou : 48 tiles
#define WS_WIOU_DW (80 * TILE_DW)   // W_iou : 48 tiles
#define WS_TOTAL_DW (128 * TILE_DW) // 262144 dwords = 1 MiB
#define WS_TOTAL_BYTES ((size_t)WS_TOTAL_DW * 4)

typedef __attribute__((ext_vector_type(16))) __bf16 v16bf;
typedef __attribute__((ext_vector_type(2)))  __bf16 v2bf;
typedef __attribute__((ext_vector_type(8)))  float  v8f;

union FragB {
    uint32_t u[8];
    uint4    q[2];
    v16bf    v;
};

// f32 pair -> packed bf16 dword (low = first element)
__device__ __forceinline__ uint32_t pk_bf16(float a, float b) {
#if defined(__gfx1250__) && __has_builtin(__builtin_amdgcn_cvt_pk_bf16_f32)
    union { v2bf v; uint32_t u; } r;
    r.v = __builtin_amdgcn_cvt_pk_bf16_f32(a, b);
    return r.u;
#else
    // single v_perm_b32: take high 16 bits of each float (truncation)
    return __builtin_amdgcn_perm(__float_as_uint(b), __float_as_uint(a),
                                 0x07060302u);
#endif
}

__device__ __forceinline__ float sigmoidf_fast(float v) {
    return 1.0f / (1.0f + __expf(-v));
}

__device__ __forceinline__ float tanhf_fast(float v) {
    float e = __expf(-2.0f * fabsf(v));
    float r = (1.0f - e) / (1.0f + e);
    return copysignf(r, v);
}

// A-fragment (16x32 bf16, M x K) from LDS tile stored as [16 rows][128 dwords]
// lane<16: row=lane,  K = kk*32 + {0..7, 16..23}
// lane>=16: row=lane-16, K = kk*32 + {8..15, 24..31}
__device__ __forceinline__ v16bf ldsA(const uint32_t* __restrict__ buf, int kk, int lane) {
    const int row = lane & 15;
    const int ch  = (lane >> 4) & 1;
    const uint32_t* p = buf + row * 128 + kk * 16 + ch * 4;
    FragB f;
    f.q[0] = *(const uint4*)(p);
    f.q[1] = *(const uint4*)(p + 8);
    return f.v;
}

// B-fragment from pre-swizzled bf16 workspace: 2 contiguous b128 per lane.
__device__ __forceinline__ v16bf ldBws(const uint32_t* __restrict__ base,
                                       int tileLocal, int kk, int lane) {
    const uint32_t* p = base + ((size_t)(tileLocal * 8 + kk) * 32 + lane) * 8;
    FragB f;
    f.q[0] = *(const uint4*)(p);
    f.q[1] = *(const uint4*)(p + 4);
    return f.v;
}

// Fallback B-fragment: gather + convert from row-major f32 W[G][256].
__device__ __forceinline__ v16bf ldBf32(const float* __restrict__ W, int colbase,
                                        int kbase, int lane) {
    const int col  = colbase + (lane & 15);
    const int coff = ((lane >> 4) & 1) * 8;
    const float* p = W + (size_t)col * 256 + kbase + coff;
    const float4 a = *(const float4*)(p);
    const float4 b = *(const float4*)(p + 4);
    const float4 c = *(const float4*)(p + 16);
    const float4 d = *(const float4*)(p + 20);
    FragB f;
    f.u[0] = pk_bf16(a.x, a.y); f.u[1] = pk_bf16(a.z, a.w);
    f.u[2] = pk_bf16(b.x, b.y); f.u[3] = pk_bf16(b.z, b.w);
    f.u[4] = pk_bf16(c.x, c.y); f.u[5] = pk_bf16(c.z, c.w);
    f.u[6] = pk_bf16(d.x, d.y); f.u[7] = pk_bf16(d.z, d.w);
    return f.v;
}

#define WMMA_BF16(A, B, C) \
    __builtin_amdgcn_wmma_f32_16x16x32_bf16(false, (A), false, (B), (short)0, (C), false, false)

// ---------------------------------------------------------------------------
// Prep kernel: convert all four weight matrices into the swizzled bf16
// B-fragment layout. One thread per output dword.
// ---------------------------------------------------------------------------
__global__ __launch_bounds__(256)
void prep_weights_kernel(const float* __restrict__ U_f,
                         const float* __restrict__ W_f,
                         const float* __restrict__ U_iou,
                         const float* __restrict__ W_iou,
                         uint32_t* __restrict__ ws)
{
    const int gid = blockIdx.x * blockDim.x + threadIdx.x;   // dword index
    if (gid >= WS_TOTAL_DW) return;
    const int frag  = gid >> 8;        // 256 dwords per fragment
    const int rem   = gid & 255;
    const int lane  = rem >> 3;
    const int e2    = rem & 7;         // dword within lane (8)
    const int tileg = frag >> 3;       // global 16-col tile index (0..127)
    const int kk    = frag & 7;

    const float* W; int tloc;
    if (tileg < 16)      { W = U_f;   tloc = tileg; }
    else if (tileg < 32) { W = W_f;   tloc = tileg - 16; }
    else if (tileg < 80) { W = U_iou; tloc = tileg - 32; }
    else                 { W = W_iou; tloc = tileg - 80; }

    const int col  = tloc * 16 + (lane & 15);
    const int coff = ((lane >> 4) & 1) * 8;
    const int e    = e2 * 2;   // element pair never straddles the 8-groups
    const int k0   = kk * 32 + (e < 8 ? coff + e : 16 + coff + (e - 8));
    const float f0 = W[(size_t)col * 256 + k0];
    const float f1 = W[(size_t)col * 256 + k0 + 1];
    ws[gid] = pk_bf16(f0, f1);
}

// ---------------------------------------------------------------------------
// Main fused kernel.
// ---------------------------------------------------------------------------
template <bool USE_WS>
__global__ __launch_bounds__(NTHREADS)
void tree_cell_kernel(const float* __restrict__ x,
                      const float* __restrict__ h_mail,
                      const float* __restrict__ c_mail,
                      const float* __restrict__ time_mail,
                      const float* __restrict__ mask,
                      const float* __restrict__ W_iou,
                      const float* __restrict__ U_iou,
                      const float* __restrict__ b_iou,
                      const float* __restrict__ U_f,
                      const float* __restrict__ W_f,
                      const float* __restrict__ b_f,
                      const uint32_t* __restrict__ ws,
                      float* __restrict__ d_out)
{
    __shared__ uint32_t xs[TILE_M * 128];      // x tile, bf16 pairs
    __shared__ uint32_t hs[2][TILE_M * 128];   // double-buffered h_m tile
    __shared__ float    ms[TILE_M * KMAIL];

    const int t    = threadIdx.x;
    const int lane = t & 31;
    const int wave = t >> 5;
    const int m0   = blockIdx.x * TILE_M;
    const int g0   = wave * 32;            // this wave's 32 output columns

    const int t2   = t & 127;              // column-pair owned by loader thread
    const int rblk = t >> 7;

    const uint32_t* wsUf   = ws + WS_UF_DW;
    const uint32_t* wsWf   = ws + WS_WF_DW;
    const uint32_t* wsUiou = ws + WS_UIOU_DW;
    const uint32_t* wsWiou = ws + WS_WIOU_DW;

    // ---- per-thread accumulators --------------------------------------------
    float htild[8][2];
#pragma unroll
    for (int i = 0; i < 8; ++i) { htild[i][0] = 0.0f; htild[i][1] = 0.0f; }

    // producer: stream h_mail/time_mail for slot k (each element read once),
    // accumulate masked h_tild, deposit bf16 h_m tile in LDS buffer (k&1).
    auto produce = [&](int k) {
        uint32_t* __restrict__ dst = hs[k & 1];
#pragma unroll
        for (int i = 0; i < 8; ++i) {
            const int r = rblk * 8 + i;
            const size_t off = ((size_t)(m0 + r) * KMAIL + k) * HDIM + 2 * t2;
            const float2 hv = *(const float2*)(h_mail + off);
            const float2 tv = *(const float2*)(time_mail + off);
            const float a0 = hv.x + tv.x;
            const float a1 = hv.y + tv.y;
            const float mv = ms[r * KMAIL + k];
            htild[i][0] += mv * a0;
            htild[i][1] += mv * a1;
            dst[r * 128 + t2] = pk_bf16(a0, a1);
        }
        if (k + 1 < KMAIL) {  // cache hints for the next slot (global_prefetch)
            const size_t noff = ((size_t)(m0 + rblk * 8) * KMAIL + (k + 1)) * HDIM + 2 * t2;
            __builtin_prefetch(h_mail + noff, 0, 0);
            __builtin_prefetch(time_mail + noff, 0, 0);
            __builtin_prefetch(c_mail + noff, 0, 0);
        }
    };

    // ---- mask tile + x tile (f32 -> bf16 LDS) + first h_m slot --------------
    ms[t] = mask[(size_t)m0 * KMAIL + t];
#pragma unroll
    for (int i = 0; i < 8; ++i) {
        const int r = rblk * 8 + i;
        const float2 v = *(const float2*)(x + (size_t)(m0 + r) * XDIM + 2 * t2);
        xs[r * 128 + t2] = pk_bf16(v.x, v.y);
    }
    produce(0);
    __syncthreads();

    // ---- preload U_f B-fragments into registers (reused for all 16 k) ------
    v16bf uf[2][8];
#pragma unroll
    for (int ct = 0; ct < 2; ++ct)
#pragma unroll
        for (int kk = 0; kk < 8; ++kk) {
            if constexpr (USE_WS)
                uf[ct][kk] = ldBws(wsUf, wave * 2 + ct, kk, lane);
            else
                uf[ct][kk] = ldBf32(U_f, g0 + ct * 16, kk * 32, lane);
        }

    // ---- xwfb = x @ W_f^T + b_f  (C-initializer for every k) ----------------
    v8f xwfb[2];
#pragma unroll
    for (int ct = 0; ct < 2; ++ct) {
        const float bv = b_f[g0 + ct * 16 + (lane & 15)];
        v8f acc;
#pragma unroll
        for (int j = 0; j < 8; ++j) acc[j] = bv;
#pragma unroll
        for (int kk = 0; kk < 8; ++kk) {
            v16bf a = ldsA(xs, kk, lane);
            v16bf b;
            if constexpr (USE_WS)
                b = ldBws(wsWf, wave * 2 + ct, kk, lane);
            else
                b = ldBf32(W_f, g0 + ct * 16, kk * 32, lane);
            acc = WMMA_BF16(a, b, acc);
        }
        xwfb[ct] = acc;
    }

    v8f cred[2];
#pragma unroll
    for (int j = 0; j < 8; ++j) { cred[0][j] = 0.0f; cred[1][j] = 0.0f; }

    const int hi = (lane >> 4) & 1;
    const int cl = lane & 15;

    // ---- main loop over the K=16 mail slots (1 barrier per slot) ------------
    for (int k = 0; k < KMAIL; ++k) {
        // kick off next slot's stream into the other LDS buffer first
        if (k + 1 < KMAIL) produce(k + 1);

        // consumer: f_pre = h_m @ U_f^T + xwfb   (per-wave column slice)
        const uint32_t* __restrict__ cur = hs[k & 1];
        v8f acc0 = xwfb[0];
        v8f acc1 = xwfb[1];
#pragma unroll
        for (int kk = 0; kk < 8; ++kk) {
            v16bf a = ldsA(cur, kk, lane);
            acc0 = WMMA_BF16(a, uf[0][kk], acc0);
            acc1 = WMMA_BF16(a, uf[1][kk], acc1);
        }

        // f = sigmoid(f_pre); c_red += mask * f * c_mail  (D-layout elementwise)
#pragma unroll
        for (int j = 0; j < 8; ++j) {
            const int row = j + hi * 8;
            const float mv = ms[row * KMAIL + k];
            const size_t off = ((size_t)(m0 + row) * KMAIL + k) * HDIM;
            const float f0 = sigmoidf_fast(acc0[j]);
            const float f1 = sigmoidf_fast(acc1[j]);
            cred[0][j] += mv * f0 * c_mail[off + g0 + cl];
            cred[1][j] += mv * f1 * c_mail[off + g0 + 16 + cl];
        }
        __syncthreads();  // produce(k+1) complete; hs[(k+1)&1] safe to consume
    }

    // ---- stash h_tild (bf16) into LDS, reusing h_m buffer 0 -----------------
#pragma unroll
    for (int i = 0; i < 8; ++i) {
        const int r = rblk * 8 + i;
        hs[0][r * 128 + t2] = pk_bf16(htild[i][0], htild[i][1]);
    }
    __syncthreads();

    // ---- iou = h_tild @ U_iou^T + x @ W_iou^T + b_iou -----------------------
    v8f iou[3][2];
#pragma unroll
    for (int p = 0; p < 3; ++p) {
#pragma unroll
        for (int ct = 0; ct < 2; ++ct) {
            const int gb = p * 256 + g0 + ct * 16;   // global column base (0..767)
            const int tl = p * 16 + wave * 2 + ct;   // local tile index (0..47)
            const float bv = b_iou[gb + (lane & 15)];
            v8f acc;
#pragma unroll
            for (int j = 0; j < 8; ++j) acc[j] = bv;
#pragma unroll
            for (int kk = 0; kk < 8; ++kk) {
                v16bf a = ldsA(hs[0], kk, lane);
                v16bf b;
                if constexpr (USE_WS)
                    b = ldBws(wsUiou, tl, kk, lane);
                else
                    b = ldBf32(U_iou, gb, kk * 32, lane);
                acc = WMMA_BF16(a, b, acc);
            }
#pragma unroll
            for (int kk = 0; kk < 8; ++kk) {
                v16bf a = ldsA(xs, kk, lane);
                v16bf b;
                if constexpr (USE_WS)
                    b = ldBws(wsWiou, tl, kk, lane);
                else
                    b = ldBf32(W_iou, gb, kk * 32, lane);
                acc = WMMA_BF16(a, b, acc);
            }
            iou[p][ct] = acc;
        }
    }

    // ---- epilogue: c = sig(i)*tanh(u) + c_red ; h = sig(o)*tanh(c) ----------
    float* __restrict__ out_h = d_out;
    float* __restrict__ out_c = d_out + (size_t)NNODES * HDIM;
#pragma unroll
    for (int ct = 0; ct < 2; ++ct) {
#pragma unroll
        for (int j = 0; j < 8; ++j) {
            const int row = j + hi * 8;
            const int col = g0 + ct * 16 + cl;
            const float iv = iou[0][ct][j];
            const float ov = iou[1][ct][j];
            const float uv = iou[2][ct][j];
            const float cc = sigmoidf_fast(iv) * tanhf_fast(uv) + cred[ct][j];
            const float hh = sigmoidf_fast(ov) * tanhf_fast(cc);
            const size_t o = (size_t)(m0 + row) * HDIM + col;
            out_h[o] = hh;
            out_c[o] = cc;
        }
    }
}

extern "C" void kernel_launch(void* const* d_in, const int* in_sizes, int n_in,
                              void* d_out, int out_size, void* d_ws, size_t ws_size,
                              hipStream_t stream) {
    (void)in_sizes; (void)n_in; (void)out_size;
    const float* x         = (const float*)d_in[0];
    const float* h_mail    = (const float*)d_in[1];
    const float* c_mail    = (const float*)d_in[2];
    const float* time_mail = (const float*)d_in[3];
    const float* mask      = (const float*)d_in[4];
    const float* W_iou     = (const float*)d_in[5];
    const float* U_iou     = (const float*)d_in[6];
    const float* b_iou     = (const float*)d_in[7];
    const float* U_f       = (const float*)d_in[8];
    const float* W_f       = (const float*)d_in[9];
    const float* b_f       = (const float*)d_in[10];

    dim3 grid(NNODES / TILE_M);   // 1250 workgroups
    dim3 block(NTHREADS);         // 8 wave32 waves

    if (ws_size >= WS_TOTAL_BYTES && d_ws != nullptr) {
        uint32_t* ws = (uint32_t*)d_ws;
        prep_weights_kernel<<<dim3(WS_TOTAL_DW / 256), dim3(256), 0, stream>>>(
            U_f, W_f, U_iou, W_iou, ws);
        tree_cell_kernel<true><<<grid, block, 0, stream>>>(
            x, h_mail, c_mail, time_mail, mask,
            W_iou, U_iou, b_iou, U_f, W_f, b_f, ws, (float*)d_out);
    } else {
        tree_cell_kernel<false><<<grid, block, 0, stream>>>(
            x, h_mail, c_mail, time_mail, mask,
            W_iou, U_iou, b_iou, U_f, W_f, b_f, nullptr, (float*)d_out);
    }
}
